// ASTEncoder_44074954392006
// MI455X (gfx1250) — matrix operand order repaired and checked by reading, not compiled, not run on hardware
//
#include <hip/hip_runtime.h>

typedef __attribute__((ext_vector_type(16))) _Float16 v16h;
typedef __attribute__((ext_vector_type(8)))  float    v8f;
typedef unsigned int u32x4 __attribute__((ext_vector_type(4)));
typedef int          i32x4 __attribute__((ext_vector_type(4)));
typedef int          i32x8 __attribute__((ext_vector_type(8)));

#define DDIM   128           // embedding dim
#define GDIM   512           // 4*D gates
#define CMAX   8             // max arity
#define TSTEPS 9             // CMAX+1
#define NCT    32            // gate column tiles (512/16)
#define NKT    4             // K tiles (128/32)
#define FRAG_HALFS (NCT*NKT*32*16)   // 65536 f16 per weight matrix

// shared memory (bytes):
//  sWih 131072 | sWhh 131072 | sX[2] 8192 | sH[2] 8192 | sBias 2048 | sNode 64 | sNC 64
#define SMEM_BYTES (131072*2 + 8192 + 8192 + 2048 + 128)

// ---------------------------------------------------------------------------
// results[i][d] = emb[type[i]][d] for i<N (covers every leaf), row N = 0 (sentinel)
__global__ void tl_init_results(const float* __restrict__ emb,
                                const int* __restrict__ types,
                                float* __restrict__ results, int Ntot) {
    int i = blockIdx.x * blockDim.x + threadIdx.x;
    int total = (Ntot + 1) * DDIM;
    if (i >= total) return;
    int node = i >> 7, d = i & 127;
    results[i] = (node < Ntot) ? emb[types[node] * DDIM + d] : 0.0f;
}

// ---------------------------------------------------------------------------
// Pack W (row-major [512,128]) into f16 WMMA B-fragments:
// frag[(ct*NKT+kt)*32 + lane][i] = W^T[kt*32 + (lane>>4)*8 + (i&7) + (i>>3)*16][ct*16 + (lane&15)]
__global__ void tl_pack_weights(const float* __restrict__ W, _Float16* __restrict__ out) {
    int t = blockIdx.x * blockDim.x + threadIdx.x;
    if (t >= NCT * NKT * 32) return;
    int lane = t & 31;
    int kt   = (t >> 5) & 3;
    int ct   = t >> 7;
    int col  = ct * 16 + (lane & 15);
    int kg   = kt * 32 + (lane >> 4) * 8;
    _Float16* dst = out + (size_t)t * 16;
#pragma unroll
    for (int i = 0; i < 16; ++i) {
        dst[i] = (_Float16)W[col * DDIM + kg + (i & 7) + ((i >> 3) << 4)];
    }
}

// ---------------------------------------------------------------------------
// One block = 16 internal nodes of one level. 256 threads = 8 waves.
// Wave w owns gate column tile ct = 8q + w for gate q in {i,f,g,o}: all four
// gates of dims [16w,16w+16) land in this wave's accumulators -> LSTM cell
// update fully in registers. One barrier per step (double-buffered sX/sH).
// Weight fragments (256 KB) arrive in LDS via the Tensor Data Mover.
__global__ __launch_bounds__(256)
void tl_lstm_level(const float* __restrict__ emb,
                   const int* __restrict__ node_types,
                   const int* __restrict__ children,
                   const int* __restrict__ num_children,
                   const int* __restrict__ level_nodes, int row_base,
                   float* __restrict__ results,
                   const _Float16* __restrict__ wih_frag,   // wih|whh contiguous
                   const float* __restrict__ b_ih,
                   const float* __restrict__ b_hh,
                   int n_internal, int Ntot) {
    extern __shared__ __align__(16) char smem[];
    _Float16* sWih  = (_Float16*)smem;                 // 65536 halfs
    _Float16* sWhh  = sWih + FRAG_HALFS;               // 65536 halfs
    _Float16* sXbuf = sWhh + FRAG_HALFS;               // 2 x 2048 halfs (A-frag of x_t)
    _Float16* sHbuf = sXbuf + 2 * 2048;                // 2 x 2048 halfs (A-frag of h)
    float*    sBias = (float*)(sHbuf + 2 * 2048);      // 512
    int*      sNode = (int*)(sBias + GDIM);            // 16
    int*      sNC   = sNode + 16;                      // 16

    const int tid  = threadIdx.x;
    const int lane = tid & 31;
    const int wave = tid >> 5;
    const int l15  = lane & 15;
    const int g16  = lane >> 4;

    // ---- TDM: DMA both weight-fragment arrays (256 KB) global -> LDS ----
    if (wave == 0) {
        unsigned lds_off       = (unsigned)(unsigned long long)(void*)sWih;
        unsigned long long ga  = (unsigned long long)(const void*)wih_frag;
        // D# group0: count=1 | lds_addr | global_addr | type=2
        u32x4 g0 = { 1u, lds_off, (unsigned)ga,
                     (unsigned)((ga >> 32) & 0x1FFFFFFull) | 0x80000000u };
        // D# group1: data_size=4B; tensor 8192x8 dwords; tile 8192x8; stride0 8192
        i32x8 g1 = { (int)0x20000,
                     (int)(8192u << 16),   // tensor_dim0[15:0] << 16
                     (int)(8u << 16),      // tensor_dim1[15:0] << 16
                     (int)(8192u << 16),   // tile_dim0 << 16
                     8,                    // tile_dim1 = 8, tile_dim2 = 0
                     8192, 0, 0 };         // tensor_dim0_stride
        // groups 2/3: dims 2..4 = 1, tiles unused
        i32x4 g2 = { 1, 1, 0, 0 };
        i32x4 g3 = { 0, (int)(1u << 16), 0, 0 };
#if __clang_major__ >= 23
        i32x8 gz8 = { 0, 0, 0, 0, 0, 0, 0, 0 };
        __builtin_amdgcn_tensor_load_to_lds(g0, g1, g2, g3, gz8, 0);
#else
        __builtin_amdgcn_tensor_load_to_lds(g0, g1, g2, g3, 0);
#endif
    }

    // ---- overlap: tile node ids, bias, h0 while the TDM streams weights ----
    if (tid < 16) {
        int idx  = blockIdx.x * 16 + tid;
        int node = (idx < n_internal) ? level_nodes[row_base + idx] : Ntot;
        if (node > Ntot) node = Ntot;
        sNode[tid] = node;
        sNC[tid]   = (node < Ntot) ? num_children[node] : 0;
    }
    for (int j = tid; j < GDIM; j += 256) sBias[j] = b_ih[j] + b_hh[j];
    for (int i = tid; i < 2048; i += 256) sHbuf[i] = (_Float16)0.0f;  // h0 = 0

    if (wave == 0) __builtin_amdgcn_s_wait_tensorcnt(0);
    __syncthreads();

    // per-lane row metadata: this lane's D rows are r + 8*(lane>>4)
    int ncr[8], noder[8];
#pragma unroll
    for (int r = 0; r < 8; ++r) {
        int rw = r + 8 * g16;
        int nd = sNode[rw];
        int nc = sNC[rw];
        noder[r] = nd;
        ncr[r]   = (nd < Ntot && nc > 0) ? nc : -1;   // -1 => never valid
    }

    // loop-invariant bias accumulator init (one v8f per gate)
    v8f binit[4];
#pragma unroll
    for (int q = 0; q < 4; ++q) {
        float b = sBias[(q * 8 + wave) * 16 + l15];
#pragma unroll
        for (int r = 0; r < 8; ++r) binit[q][r] = b;
    }

    // this lane's hidden dim (fixed): dim = 16*wave + l15
    const int dim = wave * 16 + l15;
    const int kl2  = dim & 31;
    const int gg2  = (kl2 >> 3) & 1;
    const int vv2  = ((kl2 >> 4) << 2) | ((kl2 >> 1) & 3);
    const int hoff = (((dim >> 5) * 32) + 8 * g16 + 16 * gg2) * 16 + 2 * vv2 + (kl2 & 1);

    float creg[8], hreg[8];
#pragma unroll
    for (int r = 0; r < 8; ++r) { creg[r] = 0.0f; hreg[r] = 0.0f; }

    // staging: this thread fills float4-groups j = 2*tid, 2*tid+1 (row = j>>5)
    const int srow = (2 * tid) >> 5;

    for (int t = 0; t < TSTEPS; ++t) {
        const int rb = t & 1;          // read buffers for this step
        const int wb = rb ^ 1;         // h write buffer (next step's read)
        _Float16* xb = sXbuf + rb * 2048;
        _Float16* hr = sHbuf + rb * 2048;
        _Float16* hw = sHbuf + wb * 2048 + hoff;

        // ---- stage x_t as f16 A-fragments (2 float4 groups / thread) ----
        {
            int nd = sNode[srow];
            const float* src = nullptr;
            if (nd < Ntot) {
                if (t == 0) src = emb + (size_t)node_types[nd] * DDIM;
                else        src = results + (size_t)children[nd * CMAX + (t - 1)] * DDIM;
            }
#pragma unroll
            for (int u = 0; u < 2; ++u) {
                int jj = 2 * tid + u;
                int k0 = (jj & 31) * 4;
                float4 val = make_float4(0.f, 0.f, 0.f, 0.f);
                if (src) val = *(const float4*)(src + k0);
                int kt  = k0 >> 5, kl0 = k0 & 31;
                int gg  = (kl0 >> 3) & 1;
                int vv  = ((kl0 >> 4) << 2) | ((kl0 >> 1) & 3);
                union { _Float16 h[4]; unsigned long long u64; } cv;
                cv.h[0] = (_Float16)val.x; cv.h[1] = (_Float16)val.y;
                cv.h[2] = (_Float16)val.z; cv.h[3] = (_Float16)val.w;
                *(unsigned long long*)(xb + (kt * 32 + srow + 16 * gg) * 16 + 2 * vv) = cv.u64;
            }
        }
        __syncthreads();   // the only barrier per step

        // ---- A fragments (shared by all 4 gate tiles of this wave) ----
        v16h ax[NKT], ah[NKT];
#pragma unroll
        for (int kt = 0; kt < NKT; ++kt) {
            ax[kt] = *(const v16h*)(xb + (kt * 32 + lane) * 16);
            ah[kt] = *(const v16h*)(hr + (kt * 32 + lane) * 16);
        }

        // ---- gates q = i,f,g,o : ct = 8q + wave ----
        v8f gacc[4];
#pragma unroll
        for (int q = 0; q < 4; ++q) {
            int ct = q * 8 + wave;
            v8f acc = binit[q];
#pragma unroll
            for (int kt = 0; kt < NKT; ++kt) {
                v16h bw = *(const v16h*)(sWih + ((ct * NKT + kt) * 32 + lane) * 16);
                acc = __builtin_amdgcn_wmma_f32_16x16x32_f16(false, ax[kt], false, bw,
                                                             (short)0, acc, false, false);
            }
#pragma unroll
            for (int kt = 0; kt < NKT; ++kt) {
                v16h bw = *(const v16h*)(sWhh + ((ct * NKT + kt) * 32 + lane) * 16);
                acc = __builtin_amdgcn_wmma_f32_16x16x32_f16(false, ah[kt], false, bw,
                                                             (short)0, acc, false, false);
            }
            gacc[q] = acc;
        }

        // ---- LSTM cell update in registers (masked: valid iff t <= nc) ----
#pragma unroll
        for (int r = 0; r < 8; ++r) {
            float si = 1.0f / (1.0f + __expf(-gacc[0][r]));
            float sf = 1.0f / (1.0f + __expf(-gacc[1][r]));
            float tg = tanhf(gacc[2][r]);
            float so = 1.0f / (1.0f + __expf(-gacc[3][r]));
            float cn = sf * creg[r] + si * tg;
            float hn = so * tanhf(cn);
            if (t <= ncr[r]) { creg[r] = cn; hreg[r] = hn; }
            hw[r * 16] = (_Float16)hreg[r];        // repack h into next A-frag buffer
        }
    }

    // ---- store results for real internal nodes (dim fixed per lane) ----
    float* outp = results + dim;
#pragma unroll
    for (int r = 0; r < 8; ++r) {
        if (ncr[r] >= 1) outp[(size_t)noder[r] * DDIM] = hreg[r];
    }
}

// ---------------------------------------------------------------------------
extern "C" void kernel_launch(void* const* d_in, const int* in_sizes, int n_in,
                              void* d_out, int out_size, void* d_ws, size_t ws_size,
                              hipStream_t stream) {
    const float* emb         = (const float*)d_in[0];   // [131,128]
    const float* W_ih        = (const float*)d_in[1];   // [512,128]
    const float* W_hh        = (const float*)d_in[2];   // [512,128]
    const float* b_ih        = (const float*)d_in[3];   // [512]
    const float* b_hh        = (const float*)d_in[4];   // [512]
    const int*   node_types  = (const int*)d_in[5];     // [N]
    const int*   children    = (const int*)d_in[6];     // [N,8]
    const int*   num_child   = (const int*)d_in[7];     // [N]
    const int*   level_nodes = (const int*)d_in[8];     // [L,M]

    const int N  = in_sizes[5];
    const int Cw = in_sizes[6] / N;                     // 8
    const int LM = in_sizes[8];

    // complete C-ary tree level starts: st[d+1] = st[d]*C + 1
    long st[20];
    st[0] = 0;
    for (int d = 1; d < 20; ++d) st[d] = st[d - 1] * Cw + 1;
    int L = 1;
    for (int d = 1; d < 19; ++d) if (st[d] <= (long)N - 1) L = d + 1;
    const int M = LM / L;
    const long Plast = (N >= 2) ? (long)(N - 2) / Cw : -1;  // last internal node id

    // workspace: results [(N+1),128] f32, then the two contiguous frag arrays
    float*     results  = (float*)d_ws;
    size_t     resBytes = ((size_t)(N + 1) * DDIM * sizeof(float) + 255) & ~(size_t)255;
    _Float16*  wihF     = (_Float16*)((char*)d_ws + resBytes);
    _Float16*  whhF     = wihF + FRAG_HALFS;

    (void)ws_size; (void)n_in; (void)out_size;

    hipFuncSetAttribute((const void*)tl_lstm_level,
                        hipFuncAttributeMaxDynamicSharedMemorySize, SMEM_BYTES);

    {   // leaves + sentinel
        int total = (N + 1) * DDIM;
        tl_init_results<<<(total + 255) / 256, 256, 0, stream>>>(emb, node_types, results, N);
    }
    {   // weight fragment packing
        int threads = NCT * NKT * 32;
        tl_pack_weights<<<(threads + 255) / 256, 256, 0, stream>>>(W_ih, wihF);
        tl_pack_weights<<<(threads + 255) / 256, 256, 0, stream>>>(W_hh, whhF);
    }

    // levels: deepest -> root; only internal nodes (prefix of each sorted level row)
    for (int d = L - 1; d >= 0; --d) {
        long lo = st[d];
        long nexts = st[d + 1] < (long)N ? st[d + 1] : (long)N;
        long hi = nexts - 1;
        if (lo > Plast) continue;
        long icnt = ((hi < Plast) ? hi : Plast) - lo + 1;
        if (icnt <= 0) continue;
        int row    = L - 1 - d;
        int blocks = (int)((icnt + 15) / 16);
        tl_lstm_level<<<blocks, 256, SMEM_BYTES, stream>>>(
            emb, node_types, children, num_child, level_nodes, row * M,
            results, wihF, b_ih, b_hh, (int)icnt, N);
    }

    // root encoding = results[0][0..127]
    hipMemcpyAsync(d_out, results, DDIM * sizeof(float),
                   hipMemcpyDeviceToDevice, stream);
}
